// Block_49185965473965
// MI455X (gfx1250) — compile-verified, matrix operand order checked by reading, not run on hardware
//
#include <hip/hip_runtime.h>
#include <cstdint>

// ---------------------------------------------------------------------------
// MI455X (gfx1250) decoder layer: self-attn + cross-attn + dense MoE.
// GEMMs on v_wmma_f32_16x16x32_f16 (wave32, f32 accumulate), 128x128 block
// tiles (32x64 per wave, 8 WMMAs / 32-wide K step). Flash attention keeps the
// 128MB score tensor out of HBM. Async global->LDS staging (ASYNCcnt) for the
// f16 MoE activations where available; global_prefetch for streamed weights.
// ---------------------------------------------------------------------------

#define DIM   1024
#define HEADS 16
#define DHEAD 64
#define HID   2048
#define NEXP  8
#define BB    2
#define SS    1024
#define NTOK  (BB * SS)          // 2048
#define INNER (HEADS * DHEAD)    // 1024
#define SCALE 0.125f             // DHEAD^-0.5

#if defined(__has_builtin)
# if __has_builtin(__builtin_amdgcn_global_load_async_to_lds_b128) && \
     __has_builtin(__builtin_amdgcn_s_wait_asynccnt)
#  define HAVE_ASYNC_LDS 1
# endif
#endif
#ifndef HAVE_ASYNC_LDS
# define HAVE_ASYNC_LDS 0
#endif

typedef _Float16 v16h __attribute__((ext_vector_type(16)));
typedef float    v8f  __attribute__((ext_vector_type(8)));
typedef int      v4i  __attribute__((vector_size(16)));   // b128 payload type

union Frag {
    v16h v;
    unsigned int u[8];
    _Float16 e[16];
};

__device__ __forceinline__ int frag_k(int v, int hi) {
    // 16-bit A/B fragment K index for VGPR v, lane-half hi (ISA 7.12.2)
    return (v >> 2) * 16 + hi * 8 + (v & 3) * 2;
}

// ---------------------------------------------------------------------------
// WMMA GEMM:  C[M,N] = A[M,K] * B[K,N]  (+bias[n]) (*rowscale[m])
// Block tile 128x128, 8 waves in a 4x2 grid; each wave owns a 32x64 output
// (2 A-frags x 4 B-frags -> 8 WMMAs per 32-wide K step).
// A is f32 (converted while staging) or pre-converted f16 (async DMA to LDS).
// ---------------------------------------------------------------------------
template <typename AT>
__global__ void __launch_bounds__(256)
gemm_kernel(const AT* __restrict__ A, const float* __restrict__ B,
            float* __restrict__ C, int M, int N, int K,
            const float* __restrict__ bias,
            const float* __restrict__ rowscale, int rs_stride)
{
    __shared__ _Float16 lA[128][32];   // [m][k]
    __shared__ _Float16 lB[128][32];   // [n][k]  (B staged transposed)

    const int tid  = threadIdx.x;
    const int wave = tid >> 5, lane = tid & 31;
    const int wm   = wave >> 1, wn = wave & 1;     // 4 x 2 wave grid
    const int hi   = lane >> 4, lo = lane & 15;
    const int tm   = blockIdx.y * 128, tn = blockIdx.x * 128;

    v8f c[2][4] = {};

    for (int k0 = 0; k0 < K; k0 += 32) {
        // ---- stage A tile (128 x 32) ----
#if HAVE_ASYNC_LDS
        if constexpr (sizeof(AT) == 2) {
            // f16 A: byte-exact 16B async copies, tracked by ASYNCcnt
#pragma unroll
            for (int i = 0; i < 2; ++i) {
                int lin = tid + i * 256;           // 0..511
                int r = lin >> 2, seg = (lin & 3) * 8;
                __builtin_amdgcn_global_load_async_to_lds_b128(
                    (__attribute__((address_space(1))) v4i*)
                        (A + (size_t)(tm + r) * K + k0 + seg),
                    (__attribute__((address_space(3))) v4i*)&lA[r][seg],
                    0, 0);
            }
        } else
#endif
        {
#pragma unroll
            for (int i = 0; i < 16; ++i) {
                int lin = tid + i * 256;
                int r = lin >> 5, kk = lin & 31;
                lA[r][kk] = (_Float16)A[(size_t)(tm + r) * K + k0 + kk];
            }
        }
        // ---- stage B tile (32 x 128) transposed + converted ----
#pragma unroll
        for (int i = 0; i < 16; ++i) {
            int lin = tid + i * 256;
            int kk = lin >> 7, n = lin & 127;
            lB[n][kk] = (_Float16)B[(size_t)(k0 + kk) * N + tn + n];
        }
        if (k0 + 32 < K) {                         // hint next tiles toward L2
            __builtin_prefetch(B + (size_t)(k0 + 32 + (tid >> 3)) * N + tn + (tid & 7) * 16, 0, 1);
            __builtin_prefetch(A + (size_t)(tm + (tid >> 1)) * K + k0 + 32 + (tid & 1) * 16, 0, 1);
        }
#if HAVE_ASYNC_LDS
        if constexpr (sizeof(AT) == 2) __builtin_amdgcn_s_wait_asynccnt(0);
#endif
        __syncthreads();

        Frag a[2], b[4];
#pragma unroll
        for (int v = 0; v < 8; ++v) {
            int kk = frag_k(v, hi);
#pragma unroll
            for (int i = 0; i < 2; ++i)
                a[i].u[v] = *(const unsigned int*)&lA[wm * 32 + i * 16 + lo][kk];
#pragma unroll
            for (int j = 0; j < 4; ++j)
                b[j].u[v] = *(const unsigned int*)&lB[wn * 64 + j * 16 + lo][kk];
        }
#pragma unroll
        for (int i = 0; i < 2; ++i)
#pragma unroll
            for (int j = 0; j < 4; ++j)
                c[i][j] = __builtin_amdgcn_wmma_f32_16x16x32_f16(
                    false, a[i].v, false, b[j].v, (short)0, c[i][j], false, false);
        __syncthreads();
    }

#pragma unroll
    for (int i = 0; i < 2; ++i) {
#pragma unroll
        for (int v = 0; v < 8; ++v) {
            int r = tm + wm * 32 + i * 16 + v + 8 * hi;   // C/D: M = v + 8*hi
            float sc = rowscale ? rowscale[(size_t)r * rs_stride] : 1.0f;
#pragma unroll
            for (int j = 0; j < 4; ++j) {
                int col = tn + wn * 64 + j * 16 + lo;
                float val = c[i][j][v] * sc;
                if (bias) val += bias[col];
                C[(size_t)r * N + col] = val;
            }
        }
    }
}

// ---------------------------------------------------------------------------
// Flash attention: grid (S/64, HEADS, B), block 128 (4 waves x 16 query rows).
// Per 32-key step: 4 WMMAs for Q*K^T, online softmax, 4 WMMAs for P*V.
// Q/K/V are [B, S, HEADS*DHEAD] f32.
// ---------------------------------------------------------------------------
__global__ void __launch_bounds__(128)
attn_kernel(const float* __restrict__ Q, const float* __restrict__ K,
            const float* __restrict__ V, const unsigned char* __restrict__ mask,
            float* __restrict__ O, int Skv)
{
    __shared__ _Float16 lK[32][64];        // [key][d]
    __shared__ _Float16 lVt[64][32];       // [d][key]
    __shared__ _Float16 lP[4][16][32];     // per-wave P relayout buffer

    const int tid  = threadIdx.x;
    const int wave = tid >> 5, lane = tid & 31;
    const int hi   = lane >> 4, lo = lane & 15;
    const int b = blockIdx.z, h = blockIdx.y;
    const int q0 = blockIdx.x * 64 + wave * 16;

    Frag qa0, qa1;                          // Q 16x64 as two K-chunks of 32
    {
        const float* qrow = Q + ((size_t)(b * SS + q0 + lo) * INNER) + h * DHEAD;
#pragma unroll
        for (int v = 0; v < 8; ++v) {
            int kk = frag_k(v, hi);
            float2 p0 = *(const float2*)(qrow + kk);
            float2 p1 = *(const float2*)(qrow + 32 + kk);
            qa0.e[2 * v] = (_Float16)p0.x; qa0.e[2 * v + 1] = (_Float16)p0.y;
            qa1.e[2 * v] = (_Float16)p1.x; qa1.e[2 * v + 1] = (_Float16)p1.y;
        }
    }

    float m[8], lsum[8];
#pragma unroll
    for (int v = 0; v < 8; ++v) { m[v] = -3.0e38f; lsum[v] = 0.0f; }
    v8f o0 = {}, o1 = {}, o2 = {}, o3 = {};

    for (int j0 = 0; j0 < Skv; j0 += 32) {
#pragma unroll
        for (int i = 0; i < 16; ++i) {      // stage 32 keys of K and V
            int lin = tid + i * 128;
            int key = lin >> 6, d = lin & 63;
            size_t g = ((size_t)(b * Skv + j0 + key) * INNER) + h * DHEAD + d;
            lK[key][d]  = (_Float16)K[g];
            lVt[d][key] = (_Float16)V[g];
        }
        if (j0 + 32 < Skv) {                // hint next K/V tiles toward L2
            int lin = tid * 16;
            size_t g = ((size_t)(b * Skv + j0 + 32 + (lin >> 6)) * INNER) + h * DHEAD + (lin & 63);
            __builtin_prefetch(K + g, 0, 1);
            __builtin_prefetch(V + g, 0, 1);
        }
        __syncthreads();

        // scores: S = Q * K^T  (16 x 32), two 16-key groups
        v8f s0 = {}, s1 = {};
#pragma unroll
        for (int g = 0; g < 2; ++g) {
            Frag kb;
            v8f s = {};
#pragma unroll
            for (int v = 0; v < 8; ++v)
                kb.u[v] = *(const unsigned int*)&lK[g * 16 + lo][frag_k(v, hi)];
            s = __builtin_amdgcn_wmma_f32_16x16x32_f16(false, qa0.v, false, kb.v,
                                                       (short)0, s, false, false);
#pragma unroll
            for (int v = 0; v < 8; ++v)
                kb.u[v] = *(const unsigned int*)&lK[g * 16 + lo][32 + frag_k(v, hi)];
            s = __builtin_amdgcn_wmma_f32_16x16x32_f16(false, qa1.v, false, kb.v,
                                                       (short)0, s, false, false);
            if (g == 0) s0 = s; else s1 = s;
        }
        const bool mk0 = mask[(size_t)b * Skv + j0 + lo] != 0;
        const bool mk1 = mask[(size_t)b * Skv + j0 + 16 + lo] != 0;

        // online softmax per row (row = v + 8*hi, spread over 16 lanes)
#pragma unroll
        for (int v = 0; v < 8; ++v) {
            float a0 = mk0 ? -3.0e38f : s0[v] * SCALE;
            float a1 = mk1 ? -3.0e38f : s1[v] * SCALE;
            float rm = fmaxf(a0, a1);
#pragma unroll
            for (int d = 1; d < 16; d <<= 1) rm = fmaxf(rm, __shfl_xor(rm, d, 32));
            float mn = fmaxf(m[v], rm);
            float alpha = __expf(m[v] - mn);
            float p0 = __expf(a0 - mn), p1 = __expf(a1 - mn);
            float rs = p0 + p1;
#pragma unroll
            for (int d = 1; d < 16; d <<= 1) rs += __shfl_xor(rs, d, 32);
            lsum[v] = lsum[v] * alpha + rs;
            m[v] = mn;
            o0[v] *= alpha; o1[v] *= alpha; o2[v] *= alpha; o3[v] *= alpha;
            lP[wave][v + 8 * hi][lo]      = (_Float16)p0;   // C-layout -> LDS
            lP[wave][v + 8 * hi][16 + lo] = (_Float16)p1;
        }
        asm volatile("s_wait_dscnt 0" ::: "memory");        // wave-local P ready

        Frag pa;                                            // P as A-fragment
#pragma unroll
        for (int v = 0; v < 8; ++v)
            pa.u[v] = *(const unsigned int*)&lP[wave][lo][frag_k(v, hi)];
#pragma unroll
        for (int nt = 0; nt < 4; ++nt) {                    // O += P * V
            Frag vb;
#pragma unroll
            for (int v = 0; v < 8; ++v)
                vb.u[v] = *(const unsigned int*)&lVt[nt * 16 + lo][frag_k(v, hi)];
            v8f& oo = nt == 0 ? o0 : nt == 1 ? o1 : nt == 2 ? o2 : o3;
            oo = __builtin_amdgcn_wmma_f32_16x16x32_f16(false, pa.v, false, vb.v,
                                                        (short)0, oo, false, false);
        }
        __syncthreads();
    }

#pragma unroll
    for (int v = 0; v < 8; ++v) {
        int r = q0 + v + 8 * hi;
        float inv = 1.0f / lsum[v];
        size_t base = ((size_t)(b * SS + r) * INNER) + h * DHEAD;
        O[base + lo]      = o0[v] * inv;
        O[base + 16 + lo] = o1[v] * inv;
        O[base + 32 + lo] = o2[v] * inv;
        O[base + 48 + lo] = o3[v] * inv;
    }
}

// ---------------------------------------------------------------------------
// y = LayerNorm(x + r) * g + b        one block per token row of 1024
// ---------------------------------------------------------------------------
__global__ void __launch_bounds__(256)
add_ln_kernel(const float* __restrict__ x, const float* __restrict__ r,
              const float* __restrict__ g, const float* __restrict__ bt,
              float* __restrict__ y)
{
    __shared__ float red[256];
    const int row = blockIdx.x, tid = threadIdx.x;
    const float* xr = x + (size_t)row * DIM;
    const float* rr = r + (size_t)row * DIM;
    float t[4]; float s = 0.0f;
#pragma unroll
    for (int i = 0; i < 4; ++i) { t[i] = xr[tid + i * 256] + rr[tid + i * 256]; s += t[i]; }
    red[tid] = s; __syncthreads();
    for (int st = 128; st > 0; st >>= 1) { if (tid < st) red[tid] += red[tid + st]; __syncthreads(); }
    const float mean = red[0] * (1.0f / DIM); __syncthreads();
    float s2 = 0.0f;
#pragma unroll
    for (int i = 0; i < 4; ++i) { float d = t[i] - mean; s2 += d * d; }
    red[tid] = s2; __syncthreads();
    for (int st = 128; st > 0; st >>= 1) { if (tid < st) red[tid] += red[tid + st]; __syncthreads(); }
    const float inv = rsqrtf(red[0] * (1.0f / DIM) + 1e-5f);
#pragma unroll
    for (int i = 0; i < 4; ++i) {
        int c = tid + i * 256;
        y[(size_t)row * DIM + c] = (t[i] - mean) * inv * g[c] + bt[c];
    }
}

// ---------------------------------------------------------------------------
// Router: probs = softmax(x @ gate_w); top-2 -> normalized routing weights.
// One wave (32 lanes) per token; 8 tokens per 256-thread block.
// ---------------------------------------------------------------------------
__global__ void __launch_bounds__(256)
gate_kernel(const float* __restrict__ x, const float* __restrict__ gw,
            float* __restrict__ router, float* __restrict__ wtok)
{
    const int token = blockIdx.x * 8 + (threadIdx.x >> 5);
    const int lane  = threadIdx.x & 31;
    const float* xr = x + (size_t)token * DIM;
    float acc[NEXP] = {};
    for (int d = lane; d < DIM; d += 32) {
        float xv = xr[d];
#pragma unroll
        for (int e = 0; e < NEXP; ++e) acc[e] += xv * gw[d * NEXP + e];
    }
#pragma unroll
    for (int e = 0; e < NEXP; ++e)
#pragma unroll
        for (int msk = 1; msk < 32; msk <<= 1) acc[e] += __shfl_xor(acc[e], msk, 32);

    float mx = acc[0];
#pragma unroll
    for (int e = 1; e < NEXP; ++e) mx = fmaxf(mx, acc[e]);
    float p[NEXP], sum = 0.0f;
#pragma unroll
    for (int e = 0; e < NEXP; ++e) { p[e] = __expf(acc[e] - mx); sum += p[e]; }
    const float isum = 1.0f / sum;
#pragma unroll
    for (int e = 0; e < NEXP; ++e) p[e] *= isum;

    int i1 = 0;
#pragma unroll
    for (int e = 1; e < NEXP; ++e) if (p[e] > p[i1]) i1 = e;
    int i2 = (i1 == 0) ? 1 : 0;
#pragma unroll
    for (int e = 0; e < NEXP; ++e) if (e != i1 && p[e] > p[i2]) i2 = e;
    const float wn = 1.0f / (p[i1] + p[i2]);

    if (lane == 0) {
#pragma unroll
        for (int e = 0; e < NEXP; ++e) {
            router[(size_t)token * NEXP + e] = p[e];
            wtok[(size_t)token * NEXP + e] =
                (e == i1) ? p[i1] * wn : (e == i2) ? p[i2] * wn : 0.0f;
        }
    }
}

// h = silu(h1) * h3, written as f16 so the second expert GEMM reads A directly
__global__ void swiglu_kernel(const float* __restrict__ h1,
                              const float* __restrict__ h3,
                              _Float16* __restrict__ h, size_t n)
{
    for (size_t i = (size_t)blockIdx.x * blockDim.x + threadIdx.x; i < n;
         i += (size_t)gridDim.x * blockDim.x) {
        float a = h1[i];
        h[i] = (_Float16)((a / (1.0f + __expf(-a))) * h3[i]);
    }
}

// ff += contrib; red_e[b][d] = sum_s keep(b,s) * contrib(b,s,d)
__global__ void __launch_bounds__(256)
moe_reduce_kernel(const float* __restrict__ contrib,
                  const unsigned char* __restrict__ pad,
                  const unsigned char* __restrict__ mid,
                  float* __restrict__ ff, float* __restrict__ red_e)
{
    __shared__ float buf[256];
    const int b = blockIdx.y;
    const int c = blockIdx.x * 64 + (threadIdx.x & 63);
    const int rs = threadIdx.x >> 6;
    float part = 0.0f;
    for (int s = rs; s < SS; s += 4) {
        int tok = b * SS + s;
        size_t idx = (size_t)tok * DIM + c;
        float v = contrib[idx];
        ff[idx] += v;
        if (!(pad[tok] | mid[tok])) part += v;
    }
    buf[threadIdx.x] = part; __syncthreads();
    if (rs == 0)
        red_e[(size_t)b * DIM + c] =
            buf[threadIdx.x] + buf[threadIdx.x + 64] +
            buf[threadIdx.x + 128] + buf[threadIdx.x + 192];
}

// logits[e,b] = (cumsum_e red)[b,:] . cls_w / max(#keep,1) + cls_b
__global__ void __launch_bounds__(256)
logits_kernel(const float* __restrict__ red, const unsigned char* __restrict__ pad,
              const unsigned char* __restrict__ mid,
              const float* __restrict__ cls_w, const float* __restrict__ cls_b,
              float* __restrict__ out)
{
    __shared__ float buf[256];
    const int e = blockIdx.x, b = blockIdx.y, tid = threadIdx.x;
    float cnt = 0.0f;
    for (int s = tid; s < SS; s += 256) {
        int tok = b * SS + s;
        cnt += (pad[tok] | mid[tok]) ? 0.0f : 1.0f;
    }
    buf[tid] = cnt; __syncthreads();
    for (int st = 128; st > 0; st >>= 1) { if (tid < st) buf[tid] += buf[tid + st]; __syncthreads(); }
    const float denom = fmaxf(buf[0], 1.0f); __syncthreads();

    float part = 0.0f;
    for (int d = tid; d < DIM; d += 256) {
        float c = 0.0f;
        for (int ep = 0; ep <= e; ++ep) c += red[((size_t)ep * BB + b) * DIM + d];
        part += c * cls_w[d];
    }
    buf[tid] = part; __syncthreads();
    for (int st = 128; st > 0; st >>= 1) { if (tid < st) buf[tid] += buf[tid + st]; __syncthreads(); }
    if (tid == 0) out[e * BB + b] = buf[0] / denom + cls_b[0];
}

// ---------------------------------------------------------------------------
extern "C" void kernel_launch(void* const* d_in, const int* in_sizes, int n_in,
                              void* d_out, int out_size, void* d_ws, size_t ws_size,
                              hipStream_t stream)
{
    (void)in_sizes; (void)n_in; (void)out_size; (void)ws_size;

    const float* x0  = (const float*)d_in[0];
    const float* enc = (const float*)d_in[1];
    const unsigned char* src_mask = (const unsigned char*)d_in[2];
    const unsigned char* tgt_mask = (const unsigned char*)d_in[3];
    const unsigned char* tgt_mid  = (const unsigned char*)d_in[4];
    const float* ln1_g = (const float*)d_in[5],  *ln1_b = (const float*)d_in[6];
    const float* ln2_g = (const float*)d_in[7],  *ln2_b = (const float*)d_in[8];
    const float* ln3_g = (const float*)d_in[9],  *ln3_b = (const float*)d_in[10];
    const float* sa_wq = (const float*)d_in[11], *sa_wk = (const float*)d_in[12];
    const float* sa_wv = (const float*)d_in[13], *sa_wo = (const float*)d_in[14];
    const float* sa_bo = (const float*)d_in[15];
    const float* ca_wq = (const float*)d_in[16], *ca_wk = (const float*)d_in[17];
    const float* ca_wv = (const float*)d_in[18], *ca_wo = (const float*)d_in[19];
    const float* ca_bo = (const float*)d_in[20];
    const float* gate_w = (const float*)d_in[21];
    const float* cls_w  = (const float*)d_in[22], *cls_b = (const float*)d_in[23];
    const float* moe_w1 = (const float*)d_in[24];
    const float* moe_w2 = (const float*)d_in[25];
    const float* moe_w3 = (const float*)d_in[26];

    float* out_x      = (float*)d_out;                  // [B,S,DIM]
    float* out_logits = out_x + (size_t)NTOK * DIM;     // [E,B,1]
    float* out_router = out_logits + NEXP * BB;         // [B,S,E]

    const size_t M2 = (size_t)NTOK * DIM;               // 2M floats
    float* ws   = (float*)d_ws;
    float* q    = ws + 0 * M2;
    float* kbf  = ws + 1 * M2;
    float* vbf  = ws + 2 * M2;
    float* attn = ws + 3 * M2;
    float* proj = ws + 4 * M2;
    float* x1   = ws + 5 * M2;
    float* x2   = ws + 6 * M2;
    float* ff   = ws + 7 * M2;
    // MoE-phase overlays (q..proj, x1 dead by then):
    float*    h1      = q;                    // [NTOK, HID]  (spans q+kbf)
    float*    h3      = vbf;                  // [NTOK, HID]  (spans vbf+attn)
    _Float16* h16     = (_Float16*)proj;      // [NTOK, HID] f16
    float*    contrib = x1;                   // [NTOK, DIM]
    float* wtok = ws + 8 * M2;                // [NTOK, E]
    float* red  = wtok + (size_t)NTOK * NEXP; // [E, B, DIM]

    const dim3 gP(DIM / 128, NTOK / 128);     // GEMM grid, N=1024
    const dim3 gH(HID / 128, NTOK / 128);     // GEMM grid, N=2048
    const dim3 gA(SS / 64, HEADS, BB);        // attention grid

    // ---- self attention ----
    gemm_kernel<float><<<gP, 256, 0, stream>>>(x0, sa_wq, q,   NTOK, INNER, DIM, nullptr, nullptr, 0);
    gemm_kernel<float><<<gP, 256, 0, stream>>>(x0, sa_wk, kbf, NTOK, INNER, DIM, nullptr, nullptr, 0);
    gemm_kernel<float><<<gP, 256, 0, stream>>>(x0, sa_wv, vbf, NTOK, INNER, DIM, nullptr, nullptr, 0);
    attn_kernel<<<gA, 128, 0, stream>>>(q, kbf, vbf, tgt_mask, attn, SS);
    gemm_kernel<float><<<gP, 256, 0, stream>>>(attn, sa_wo, proj, NTOK, DIM, INNER, sa_bo, nullptr, 0);
    add_ln_kernel<<<NTOK, 256, 0, stream>>>(x0, proj, ln1_g, ln1_b, x1);

    // ---- cross attention ----
    gemm_kernel<float><<<gP, 256, 0, stream>>>(x1,  ca_wq, q,   NTOK, INNER, DIM, nullptr, nullptr, 0);
    gemm_kernel<float><<<gP, 256, 0, stream>>>(enc, ca_wk, kbf, NTOK, INNER, DIM, nullptr, nullptr, 0);
    gemm_kernel<float><<<gP, 256, 0, stream>>>(enc, ca_wv, vbf, NTOK, INNER, DIM, nullptr, nullptr, 0);
    attn_kernel<<<gA, 128, 0, stream>>>(q, kbf, vbf, src_mask, attn, SS);
    gemm_kernel<float><<<gP, 256, 0, stream>>>(attn, ca_wo, proj, NTOK, DIM, INNER, ca_bo, nullptr, 0);
    add_ln_kernel<<<NTOK, 256, 0, stream>>>(x1, proj, ln2_g, ln2_b, x2);

    // ---- MoE ----
    gate_kernel<<<NTOK / 8, 256, 0, stream>>>(x2, gate_w, out_router, wtok);
    (void)hipMemsetAsync(ff, 0, M2 * sizeof(float), stream);

    for (int e = 0; e < NEXP; ++e) {
        const float* w1e = moe_w1 + (size_t)e * DIM * HID;
        const float* w2e = moe_w2 + (size_t)e * HID * DIM;
        const float* w3e = moe_w3 + (size_t)e * DIM * HID;
        gemm_kernel<float><<<gH, 256, 0, stream>>>(x2, w1e, h1, NTOK, HID, DIM, nullptr, nullptr, 0);
        gemm_kernel<float><<<gH, 256, 0, stream>>>(x2, w3e, h3, NTOK, HID, DIM, nullptr, nullptr, 0);
        swiglu_kernel<<<4096, 256, 0, stream>>>(h1, h3, h16, (size_t)NTOK * HID);
        gemm_kernel<_Float16><<<gP, 256, 0, stream>>>(h16, w2e, contrib, NTOK, DIM, HID,
                                                      nullptr, wtok + e, NEXP);
        moe_reduce_kernel<<<dim3(DIM / 64, BB), 256, 0, stream>>>(
            contrib, tgt_mask, tgt_mid, ff, red + (size_t)e * BB * DIM);
    }
    logits_kernel<<<dim3(NEXP, BB), 256, 0, stream>>>(red, tgt_mask, tgt_mid,
                                                      cls_w, cls_b, out_logits);
    add_ln_kernel<<<NTOK, 256, 0, stream>>>(x2, ff, ln3_g, ln3_b, out_x);
}
